// MPNN_1477468750486
// MI455X (gfx1250) — matrix operand order repaired and checked by reading, not compiled, not run on hardware
//
#include <hip/hip_runtime.h>
#include <hip/hip_bf16.h>
#include <cmath>

// MPNN (NNConv + GRU x3, Set2Set x3, readout) for MI455X / gfx1250.
// Strategy: never materialize per-edge [64x64] transforms (800MB). Instead
// per-node T[u] = einsum(out[u], nn2_w) via WMMA (16-node blocks), messages
// H_u[19x128] @ T[u][128x64] via WMMA, fused in one kernel with T staged in
// 64KB LDS across 4 k-phases. All GEMMs use v_wmma_f32_16x16x32_f16.

#define DIM     64
#define NGRAPHS 128
#define NPER    20
#define NNODES  (NGRAPHS*NPER)        // 2560
#define EPER    (NPER*(NPER-1))       // 380
#define NEDGES  (NGRAPHS*EPER)        // 48640
#define H1DIM   128
#define KH      32                    // k values staged in LDS per phase

typedef __attribute__((ext_vector_type(16))) _Float16 v16h;
typedef __attribute__((ext_vector_type(8)))  float    v8f;

union Frag16 { v16h v; unsigned u[8]; };

__device__ __forceinline__ v8f wmma_f16(const Frag16& a, const Frag16& b, v8f c) {
  return __builtin_amdgcn_wmma_f32_16x16x32_f16(false, a.v, false, b.v,
                                                (short)0, c, false, false);
}
// A-matrix (16x32 f16) K offset for VGPR v, lane-half hf  (ISA 7.12.2)
__device__ __forceinline__ int a_koff(int v, int hf) {
  return ((v & 4) << 2) + hf * 8 + ((v & 3) << 1);
}
// B-matrix (32x16 f16) K row for VGPR v, lane-half hf
__device__ __forceinline__ int b_koff(int v, int hf) { return ((v & 4) << 2) + hf * 8 + ((v & 3) << 1); }

__device__ __forceinline__ float sigf(float x) { return 1.f / (1.f + expf(-x)); }

// ---------------- weight conversion ----------------
// nn2t[k][o][i] = nn2_w[k][i*64+o]   (f16, i contiguous for B-frag dword loads)
__global__ void k_cvt_nn2(const float* __restrict__ nn2_w, _Float16* __restrict__ nn2t) {
  int idx = blockIdx.x * blockDim.x + threadIdx.x;
  if (idx >= 128 * 4096) return;
  int k = idx >> 12, rem = idx & 4095, i = rem >> 6, o = rem & 63;
  nn2t[(k * 64 + o) * 64 + i] = (_Float16)nn2_w[idx];
}
// rootT[o][i]=root_w[i][o]; BmatT[o][i]=nn2_b[i*64+o]; wih/whh direct copies ([out][in])
__global__ void k_cvt_small(const float* __restrict__ root_w, const float* __restrict__ nn2_b,
                            const float* __restrict__ w_ih, const float* __restrict__ w_hh,
                            _Float16* __restrict__ rootT, _Float16* __restrict__ BmatT,
                            _Float16* __restrict__ wih_h, _Float16* __restrict__ whh_h) {
  int idx = blockIdx.x * blockDim.x + threadIdx.x;
  if (idx < 4096) {
    int i = idx >> 6, o = idx & 63;
    rootT[o * 64 + i] = (_Float16)root_w[i * 64 + o];
  } else if (idx < 8192) {
    int t = idx - 4096; int i = t >> 6, o = t & 63;
    BmatT[o * 64 + i] = (_Float16)nn2_b[i * 64 + o];
  } else if (idx < 8192 + 12288) {
    int t = idx - 8192; wih_h[t] = (_Float16)w_ih[t];
  } else if (idx < 8192 + 24576) {
    int t = idx - 8192 - 12288; whh_h[t] = (_Float16)w_hh[t];
  }
}

// ---------------- tiny input layers ----------------
__global__ void k_h1(const float* __restrict__ ea, const float* __restrict__ w,
                     const float* __restrict__ b, _Float16* __restrict__ h1h) {
  int idx = blockIdx.x * blockDim.x + threadIdx.x;
  if (idx >= NEDGES * H1DIM) return;
  int e = idx >> 7, k = idx & 127;
  float s = b[k];
#pragma unroll
  for (int f = 0; f < 5; ++f) s += ea[e * 5 + f] * w[f * H1DIM + k];
  h1h[idx] = (_Float16)fmaxf(s, 0.f);
}

__global__ void k_lin0(const float* __restrict__ x, const float* __restrict__ w,
                       const float* __restrict__ b, float* __restrict__ out,
                       _Float16* __restrict__ outh) {
  int idx = blockIdx.x * blockDim.x + threadIdx.x;
  if (idx >= NNODES * 64) return;
  int n = idx >> 6, o = idx & 63;
  float s = b[o];
#pragma unroll
  for (int f = 0; f < 11; ++f) s += x[n * 11 + f] * w[f * 64 + o];
  s = fmaxf(s, 0.f);
  out[idx] = s; outh[idx] = (_Float16)s;
}

// ---------------- generic WMMA GEMM: C[M x Cc] = A[M x 64](f16) @ W[Cc x 64]^T ----
__global__ __launch_bounds__(256) void k_gemm64(const _Float16* __restrict__ A,
                                                const _Float16* __restrict__ W,
                                                float* __restrict__ C, int Cc) {
  int colTiles = Cc >> 4;
  int tile = blockIdx.x * 8 + (threadIdx.x >> 5);
  int rt = tile / colTiles, ct = tile - rt * colTiles;
  int l = threadIdx.x & 31, n = l & 15, hf = l >> 4;
  v8f c = {};
#pragma unroll
  for (int ks = 0; ks < 2; ++ks) {
    Frag16 a, b;
#pragma unroll
    for (int v = 0; v < 8; ++v)
      a.u[v] = *(const unsigned*)(A + (rt * 16 + n) * 64 + ks * 32 + a_koff(v, hf));
#pragma unroll
    for (int v = 0; v < 8; ++v)
      b.u[v] = *(const unsigned*)(W + (ct * 16 + n) * 64 + ks * 32 + b_koff(v, hf));
    c = wmma_f16(a, b, c);
  }
#pragma unroll
  for (int r = 0; r < 8; ++r)
    C[(rt * 16 + r + 8 * hf) * Cc + ct * 16 + n] = c[r];
}

// ---------------- fused NNConv kernel: T in LDS + per-edge messages ----------
__global__ __launch_bounds__(256) void k_conv(const _Float16* __restrict__ outh,
                                              const _Float16* __restrict__ nn2t,
                                              const _Float16* __restrict__ h1h,
                                              float* __restrict__ msg) {
  __shared__ _Float16 Th[16 * 64 * KH];   // Th[node][o][kk], 64KB
  const int u0 = blockIdx.x * 16;
  const int tid = threadIdx.x, wave = tid >> 5, l = tid & 31, n = l & 15, hf = l >> 4;

  // A for T-GEMM: 16 node rows x 64 i  (same for all k)
  Frag16 A1[2];
#pragma unroll
  for (int ksi = 0; ksi < 2; ++ksi)
#pragma unroll
    for (int v = 0; v < 8; ++v)
      A1[ksi].u[v] = *(const unsigned*)(outh + (u0 + n) * 64 + ksi * 32 + a_koff(v, hf));

  // phase-2 geometry: this wave owns 2 source nodes
  int ug0 = u0 + wave * 2, g0 = ug0 / 20, lu0 = ug0 - g0 * 20;
  int ug1 = ug0 + 1,       g1 = ug1 / 20, lu1 = ug1 - g1 * 20;
  long eb[2] = { (long)g0 * EPER + (long)lu0 * 19, (long)g1 * EPER + (long)lu1 * 19 };

  v8f zero = {};
  v8f acc[2][2][4];
#pragma unroll
  for (int s = 0; s < 2; ++s)
#pragma unroll
    for (int mt = 0; mt < 2; ++mt)
#pragma unroll
      for (int nt = 0; nt < 4; ++nt) acc[s][mt][nt] = zero;

  for (int ph = 0; ph < H1DIM / KH; ++ph) {
    // ---- phase 1: T[u][k][o] for k in [ph*KH, ph*KH+KH), all 16 nodes ----
#pragma unroll
    for (int t = 0; t < KH / 8; ++t) {
      int kk = wave * (KH / 8) + t;
      int k = ph * KH + kk;
      const _Float16* Bk = nn2t + k * 4096;
#pragma unroll
      for (int nt = 0; nt < 4; ++nt) {
        v8f c = {};
#pragma unroll
        for (int ksi = 0; ksi < 2; ++ksi) {
          Frag16 B;
#pragma unroll
          for (int v = 0; v < 8; ++v)
            B.u[v] = *(const unsigned*)(Bk + (nt * 16 + n) * 64 + ksi * 32 + b_koff(v, hf));
          c = wmma_f16(A1[ksi], B, c);
        }
#pragma unroll
        for (int r = 0; r < 8; ++r)
          Th[((r + 8 * hf) * 64 + nt * 16 + n) * KH + kk] = (_Float16)c[r];
      }
    }
    __syncthreads();
    // ---- phase 2: accumulate msg += H_u[32x32] @ T[u][32x64] ----
#pragma unroll
    for (int s = 0; s < 2; ++s) {
      int ul = wave * 2 + s;
#pragma unroll
      for (int mt = 0; mt < 2; ++mt) {
        int j = mt * 16 + n; if (j > 18) j = 18;    // clamp pad rows (ignored at store)
        const _Float16* Arow = h1h + (eb[s] + j) * H1DIM + ph * KH;
        Frag16 Af;
#pragma unroll
        for (int v = 0; v < 8; ++v)
          Af.u[v] = *(const unsigned*)(Arow + a_koff(v, hf));
#pragma unroll
        for (int nt = 0; nt < 4; ++nt) {
          Frag16 Bf;
#pragma unroll
          for (int v = 0; v < 8; ++v)
            Bf.u[v] = *(const unsigned*)(&Th[(ul * 64 + nt * 16 + n) * KH + b_koff(v, hf)]);
          acc[s][mt][nt] = wmma_f16(Af, Bf, acc[s][mt][nt]);
        }
      }
    }
    __syncthreads();
  }
  // ---- store messages (rows j>=19 are padding) ----
#pragma unroll
  for (int s = 0; s < 2; ++s)
#pragma unroll
    for (int mt = 0; mt < 2; ++mt)
#pragma unroll
      for (int nt = 0; nt < 4; ++nt) {
        int o = nt * 16 + n;
#pragma unroll
        for (int r = 0; r < 8; ++r) {
          int j = mt * 16 + r + 8 * hf;
          if (j < 19) msg[(eb[s] + j) * 64 + o] = acc[s][mt][nt][r];
        }
      }
}

// ---------------- aggregation: segment-mean + bias-term + root + relu ---------
__global__ void k_agg(const float* __restrict__ msg, const float* __restrict__ bb,
                      const float* __restrict__ rootterm, const float* __restrict__ conv_b,
                      float* __restrict__ m, _Float16* __restrict__ mh) {
  int g = blockIdx.x, tid = threadIdx.x;
  __shared__ float bbs[64];
  if (tid < 64) {
    float s = 0.f;
    for (int v = 0; v < NPER; ++v) s += bb[(g * NPER + v) * 64 + tid];
    bbs[tid] = s;
  }
  __syncthreads();
  long mb = (long)g * EPER * 64;
  for (int idx = tid; idx < NPER * 64; idx += blockDim.x) {
    int v = idx >> 6, o = idx & 63;
    int node = g * NPER + v;
    float s = bbs[o] - bb[node * 64 + o];
    for (int r = 0; r < NPER; ++r) {
      if (r == v) continue;
      int e = r * 19 + (v < r ? v : v - 1);
      s += msg[mb + (long)e * 64 + o];
    }
    float val = s * (1.f / 19.f) + rootterm[node * 64 + o] + conv_b[o];
    val = fmaxf(val, 0.f);
    m[node * 64 + o] = val;
    mh[node * 64 + o] = (_Float16)val;
  }
}

// ---------------- GRU gate nonlinearity + state update ----------------
__global__ void k_gru(const float* __restrict__ gi, const float* __restrict__ gh,
                      const float* __restrict__ b_ih, const float* __restrict__ b_hh,
                      float* __restrict__ out, _Float16* __restrict__ outh) {
  int idx = blockIdx.x * blockDim.x + threadIdx.x;
  if (idx >= NNODES * 64) return;
  int nn = idx >> 6, o = idx & 63;
  const float* gin = gi + nn * 192;
  const float* ghn = gh + nn * 192;
  float r = sigf(gin[o] + b_ih[o] + ghn[o] + b_hh[o]);
  float z = sigf(gin[64 + o] + b_ih[64 + o] + ghn[64 + o] + b_hh[64 + o]);
  float nv = tanhf(gin[128 + o] + b_ih[128 + o] + r * (ghn[128 + o] + b_hh[128 + o]));
  float h = (1.f - z) * nv + z * out[idx];
  out[idx] = h; outh[idx] = (_Float16)h;
}

// ---------------- Set2Set ----------------
__global__ void k_zero(float* p, int nelem) {
  int i = blockIdx.x * blockDim.x + threadIdx.x;
  if (i < nelem) p[i] = 0.f;
}

__global__ void k_lstm(const float* __restrict__ w_ih, const float* __restrict__ w_hh,
                       const float* __restrict__ b_ih, const float* __restrict__ b_hh,
                       float* __restrict__ qstar, float* __restrict__ hl,
                       float* __restrict__ cl) {
  int g = blockIdx.x, j = threadIdx.x;  // 256 threads = 4*DIM gates
  __shared__ float gates[256];
  float s = b_ih[j] + b_hh[j];
  for (int i = 0; i < 128; ++i) s += qstar[g * 128 + i] * w_ih[j * 128 + i];
  for (int i = 0; i < 64; ++i)  s += hl[g * 64 + i] * w_hh[j * 64 + i];
  gates[j] = s;
  __syncthreads();
  if (j < 64) {
    float ig = gates[j], fg = gates[64 + j], gg = gates[128 + j], og = gates[192 + j];
    float c = sigf(fg) * cl[g * 64 + j] + sigf(ig) * tanhf(gg);
    float hv = sigf(og) * tanhf(c);
    cl[g * 64 + j] = c; hl[g * 64 + j] = hv;
    qstar[g * 128 + j] = hv;               // q half of next q_star
  }
}

__global__ void k_attn(const float* __restrict__ out, const float* __restrict__ hl,
                       float* __restrict__ qstar) {
  int g = blockIdx.x, tid = threadIdx.x;   // 64 threads
  __shared__ float ee[NPER];
  __shared__ float inv_s;
  if (tid < NPER) {
    float s = 0.f;
    for (int o = 0; o < 64; ++o) s += out[(g * NPER + tid) * 64 + o] * hl[g * 64 + o];
    ee[tid] = s;
  }
  __syncthreads();
  if (tid == 0) {
    float mx = ee[0];
    for (int i = 1; i < NPER; ++i) mx = fmaxf(mx, ee[i]);
    float sm = 0.f;
    for (int i = 0; i < NPER; ++i) { ee[i] = expf(ee[i] - mx); sm += ee[i]; }
    inv_s = 1.f / sm;
  }
  __syncthreads();
  float rv = 0.f;
  for (int v = 0; v < NPER; ++v) rv += ee[v] * out[(g * NPER + v) * 64 + tid];
  qstar[g * 128 + 64 + tid] = rv * inv_s;
}

__global__ void k_readout(const float* __restrict__ qstar, const float* __restrict__ w1,
                          const float* __restrict__ b1, const float* __restrict__ w2,
                          const float* __restrict__ b2, float* __restrict__ y) {
  int g = blockIdx.x, o = threadIdx.x;     // 64 threads
  __shared__ float hbuf[64];
  float s = b1[o];
  for (int i = 0; i < 128; ++i) s += qstar[g * 128 + i] * w1[i * 64 + o];
  hbuf[o] = fmaxf(s, 0.f);
  __syncthreads();
  if (o == 0) {
    float acc = b2[0];
    for (int i = 0; i < 64; ++i) acc += hbuf[i] * w2[i];
    y[g] = acc;
  }
}

// ---------------- driver ----------------
extern "C" void kernel_launch(void* const* d_in, const int* in_sizes, int n_in,
                              void* d_out, int out_size, void* d_ws, size_t ws_size,
                              hipStream_t stream) {
  const float* x        = (const float*)d_in[0];
  const float* edge_attr= (const float*)d_in[2];
  const float* lin0_w   = (const float*)d_in[4];
  const float* lin0_b   = (const float*)d_in[5];
  const float* nn1_w    = (const float*)d_in[6];
  const float* nn1_b    = (const float*)d_in[7];
  const float* nn2_w    = (const float*)d_in[8];
  const float* nn2_b    = (const float*)d_in[9];
  const float* root_w   = (const float*)d_in[10];
  const float* conv_b   = (const float*)d_in[11];
  const float* gru_w_ih = (const float*)d_in[12];
  const float* gru_w_hh = (const float*)d_in[13];
  const float* gru_b_ih = (const float*)d_in[14];
  const float* gru_b_hh = (const float*)d_in[15];
  const float* lstm_w_ih= (const float*)d_in[16];
  const float* lstm_w_hh= (const float*)d_in[17];
  const float* lstm_b_ih= (const float*)d_in[18];
  const float* lstm_b_hh= (const float*)d_in[19];
  const float* lin1_w   = (const float*)d_in[20];
  const float* lin1_b   = (const float*)d_in[21];
  const float* lin2_w   = (const float*)d_in[22];
  const float* lin2_b   = (const float*)d_in[23];
  float* yout = (float*)d_out;

  char* wsp = (char*)d_ws;
  auto alloc = [&](size_t bytes) -> char* {
    char* p = wsp; wsp += (bytes + 255) & ~(size_t)255; return p;
  };
  _Float16* nn2t  = (_Float16*)alloc((size_t)128 * 4096 * 2);
  _Float16* rootT = (_Float16*)alloc(4096 * 2);
  _Float16* BmatT = (_Float16*)alloc(4096 * 2);
  _Float16* wih_h = (_Float16*)alloc(12288 * 2);
  _Float16* whh_h = (_Float16*)alloc(12288 * 2);
  _Float16* h1h   = (_Float16*)alloc((size_t)NEDGES * H1DIM * 2);
  float*    out   = (float*)   alloc((size_t)NNODES * 64 * 4);
  _Float16* outh  = (_Float16*)alloc((size_t)NNODES * 64 * 2);
  float*    bb    = (float*)   alloc((size_t)NNODES * 64 * 4);
  float*    rterm = (float*)   alloc((size_t)NNODES * 64 * 4);
  float*    msg   = (float*)   alloc((size_t)NEDGES * 64 * 4);
  float*    mbuf  = (float*)   alloc((size_t)NNODES * 64 * 4);
  _Float16* mh    = (_Float16*)alloc((size_t)NNODES * 64 * 2);
  float*    gi    = (float*)   alloc((size_t)NNODES * 192 * 4);
  float*    gh    = (float*)   alloc((size_t)NNODES * 192 * 4);
  float*    qstar = (float*)   alloc((size_t)NGRAPHS * 128 * 4);
  float*    hl    = (float*)   alloc((size_t)NGRAPHS * 64 * 4);
  float*    cl    = (float*)   alloc((size_t)NGRAPHS * 64 * 4);

  // one-time conversions / input layers
  k_cvt_nn2<<<(128 * 4096 + 255) / 256, 256, 0, stream>>>(nn2_w, nn2t);
  k_cvt_small<<<(8192 + 24576 + 255) / 256, 256, 0, stream>>>(
      root_w, nn2_b, gru_w_ih, gru_w_hh, rootT, BmatT, wih_h, whh_h);
  k_h1<<<(NEDGES * H1DIM + 255) / 256, 256, 0, stream>>>(edge_attr, nn1_w, nn1_b, h1h);
  k_lin0<<<(NNODES * 64 + 255) / 256, 256, 0, stream>>>(x, lin0_w, lin0_b, out, outh);

  // 3 message-passing layers
  for (int layer = 0; layer < 3; ++layer) {
    k_gemm64<<<(NNODES / 16) * 4 / 8, 256, 0, stream>>>(outh, BmatT, bb, 64);
    k_gemm64<<<(NNODES / 16) * 4 / 8, 256, 0, stream>>>(outh, rootT, rterm, 64);
    k_conv<<<NNODES / 16, 256, 0, stream>>>(outh, nn2t, h1h, msg);
    k_agg<<<NGRAPHS, 256, 0, stream>>>(msg, bb, rterm, conv_b, mbuf, mh);
    k_gemm64<<<(NNODES / 16) * 12 / 8, 256, 0, stream>>>(mh, wih_h, gi, 192);
    k_gemm64<<<(NNODES / 16) * 12 / 8, 256, 0, stream>>>(outh, whh_h, gh, 192);
    k_gru<<<(NNODES * 64 + 255) / 256, 256, 0, stream>>>(gi, gh, gru_b_ih, gru_b_hh, out, outh);
  }

  // Set2Set (3 steps) + readout
  k_zero<<<(NGRAPHS * 128 + 255) / 256, 256, 0, stream>>>(qstar, NGRAPHS * 128);
  k_zero<<<(NGRAPHS * 64 + 255) / 256, 256, 0, stream>>>(hl, NGRAPHS * 64);
  k_zero<<<(NGRAPHS * 64 + 255) / 256, 256, 0, stream>>>(cl, NGRAPHS * 64);
  for (int step = 0; step < 3; ++step) {
    k_lstm<<<NGRAPHS, 256, 0, stream>>>(lstm_w_ih, lstm_w_hh, lstm_b_ih, lstm_b_hh,
                                        qstar, hl, cl);
    k_attn<<<NGRAPHS, 64, 0, stream>>>(out, hl, qstar);
  }
  k_readout<<<NGRAPHS, 64, 0, stream>>>(qstar, lin1_w, lin1_b, lin2_w, lin2_b, yout);
}
